// CarlaRNNPPOMG_11802570129994
// MI455X (gfx1250) — matrix operand order repaired and checked by reading, not compiled, not run on hardware
//
#include <hip/hip_runtime.h>
#include <hip/hip_bf16.h>

// ---------------------------------------------------------------------------
// CarlaRNNPPO forward for MI455X (gfx1250, wave32, WMMA).
// All dense GEMMs use v_wmma_f32_16x16x32_bf16 (bf16 in, f32 accumulate).
// This revision forces address_space(1) on all fragment loads so the inner
// loop emits global_load_b128 (loadcnt only) instead of flat_load_b128
// (loadcnt+dscnt, aperture-checked generic path).
// ---------------------------------------------------------------------------

typedef __bf16 bf16;
typedef bf16  v16bf __attribute__((ext_vector_type(16)));
typedef bf16  v8bf  __attribute__((ext_vector_type(8)));
typedef float v8f   __attribute__((ext_vector_type(8)));

#define GAS __attribute__((address_space(1)))
typedef GAS const bf16*  gbf;
typedef GAS const v8bf*  gv8bf;
typedef GAS const v16bf* gv16bf;

#define T_    128
#define B_    256
#define S_    256
#define H_    512
#define M_    1024
#define G4H   2048   // 4*H
#define EGOP  32     // ego K padded from 12 -> 32 (one WMMA K-step)

static __device__ __forceinline__ gbf as_global(const bf16* p) {
  return (gbf)(unsigned long long)p;
}

// --- A-fragment loader: 16x32 bf16, contiguous K per lane half ------------------------------
// lane m = lane&15; lanes<16 hold K in [0..15], lanes>=16 K [16..31]
static __device__ __forceinline__ v16bf load_afrag(gbf row, int k0, int base0) {
  v8bf lo = *(gv8bf)(row + k0 + base0);
  v8bf hi = *(gv8bf)(row + k0 + base0 + 8);
  v16bf r;
#pragma unroll
  for (int i = 0; i < 8; ++i) { r[i] = lo[i]; r[i + 8] = hi[i]; }
  return r;
}

// B-fragment: 32x16 bf16 = W row segment (contiguous 16 K values per lane)
static __device__ __forceinline__ v16bf load_bfrag(gbf B, int ldb, int n, int kk) {
  return *(gv16bf)(B + (size_t)n * ldb + kk);
}

// ---------------------------------------------------------------------------
// Generic WMMA GEMM:  out[m][n] = act( sum_k A[m][k]*B[n][k] + bias )
//   mode 0: A rows from A0 (k<K0, stride lda0) then A1 (k>=K0, stride lda1)
//   mode 1: ragged gather; row -> rowmap (b<<16|t); k<K0 from state_bf[t][b][k],
//           k>=K0 from hs[t][b][k-K0]   (A0=state_bf, A1=hs)
//   mode 2: A rows from A0 only (K1 == 0)
// B is row-major W[n][k]: B0 covers k<K0 (ldb0), B1 covers k>=K0 (ldb1).
// Block tile 128x64, 8 waves (4x2), wave tile 32x32.
// ---------------------------------------------------------------------------
__global__ __launch_bounds__(256)
void wmma_gemm_kernel(int mode,
                      const bf16* __restrict__ A0, int lda0, int K0,
                      const bf16* __restrict__ A1, int lda1, int K1,
                      const bf16* __restrict__ B0, int ldb0,
                      const bf16* __restrict__ B1, int ldb1,
                      const float* __restrict__ bias0,
                      const float* __restrict__ bias1,
                      float* __restrict__ outF, bf16* __restrict__ outH, int ldo,
                      int act, const int* __restrict__ rowmap)
{
  const int lane  = threadIdx.x & 31;
  const int wid   = threadIdx.x >> 5;
  const int wm    = wid & 3;
  const int wn    = wid >> 2;
  const int tileM = blockIdx.x * 128 + wm * 32;
  const int tileN = blockIdx.y * 64  + wn * 32;
  const int mlane = lane & 15;
  const int base0 = (lane >> 4) * 16;   // A K-half select
  const int koffB = (lane >> 4) * 16;   // B K-half select

  gbf aseg0[2];
  gbf aseg1[2];
#pragma unroll
  for (int f = 0; f < 2; ++f) {
    int r = tileM + f * 16 + mlane;
    if (mode == 1) {
      int bm = rowmap[r];
      int b  = bm >> 16;
      int t  = bm & 0xffff;
      aseg0[f] = as_global(A0) + (size_t)t * (B_ * S_) + (size_t)b * S_;  // state_bf16
      aseg1[f] = as_global(A1) + (size_t)t * (B_ * H_) + (size_t)b * H_;  // hs bf16
    } else {
      aseg0[f] = as_global(A0) + (size_t)r * lda0;
      aseg1[f] = as_global(A1) + (size_t)r * lda1;   // only used when K1 > 0
    }
  }
  gbf Bg0 = as_global(B0);
  gbf Bg1 = as_global(B1);

  v8f zero;
#pragma unroll
  for (int e = 0; e < 8; ++e) zero[e] = 0.f;
  v8f acc[2][2];
#pragma unroll
  for (int i = 0; i < 2; ++i)
#pragma unroll
    for (int j = 0; j < 2; ++j) acc[i][j] = zero;

  const int n0 = tileN + mlane;
  const int n1 = tileN + 16 + mlane;

  // --- K segment 0 ---
  for (int k0 = 0; k0 < K0; k0 += 32) {
    v16bf a0 = load_afrag(aseg0[0], k0, base0);
    v16bf a1 = load_afrag(aseg0[1], k0, base0);
    v16bf b0 = load_bfrag(Bg0, ldb0, n0, k0 + koffB);
    v16bf b1 = load_bfrag(Bg0, ldb0, n1, k0 + koffB);
    acc[0][0] = __builtin_amdgcn_wmma_f32_16x16x32_bf16(false, a0, false, b0, (short)0, acc[0][0], false, false);
    acc[0][1] = __builtin_amdgcn_wmma_f32_16x16x32_bf16(false, a0, false, b1, (short)0, acc[0][1], false, false);
    acc[1][0] = __builtin_amdgcn_wmma_f32_16x16x32_bf16(false, a1, false, b0, (short)0, acc[1][0], false, false);
    acc[1][1] = __builtin_amdgcn_wmma_f32_16x16x32_bf16(false, a1, false, b1, (short)0, acc[1][1], false, false);
  }
  // --- K segment 1 ---
  for (int k1 = 0; k1 < K1; k1 += 32) {
    v16bf a0 = load_afrag(aseg1[0], k1, base0);
    v16bf a1 = load_afrag(aseg1[1], k1, base0);
    v16bf b0 = load_bfrag(Bg1, ldb1, n0, k1 + koffB);
    v16bf b1 = load_bfrag(Bg1, ldb1, n1, k1 + koffB);
    acc[0][0] = __builtin_amdgcn_wmma_f32_16x16x32_bf16(false, a0, false, b0, (short)0, acc[0][0], false, false);
    acc[0][1] = __builtin_amdgcn_wmma_f32_16x16x32_bf16(false, a0, false, b1, (short)0, acc[0][1], false, false);
    acc[1][0] = __builtin_amdgcn_wmma_f32_16x16x32_bf16(false, a1, false, b0, (short)0, acc[1][0], false, false);
    acc[1][1] = __builtin_amdgcn_wmma_f32_16x16x32_bf16(false, a1, false, b1, (short)0, acc[1][1], false, false);
  }

  // --- epilogue: C/D layout: VGPR r -> row r (+8 for upper lanes), col lane&15
  GAS float* outFg = (GAS float*)(unsigned long long)outF;
  GAS bf16*  outHg = (GAS bf16*)(unsigned long long)outH;
  const int rbase = (lane >> 4) * 8;
#pragma unroll
  for (int fm = 0; fm < 2; ++fm) {
#pragma unroll
    for (int fn = 0; fn < 2; ++fn) {
      int n = tileN + fn * 16 + mlane;
      float bs = 0.f;
      if (bias0) bs += bias0[n];
      if (bias1) bs += bias1[n];
      v8f c = acc[fm][fn];
#pragma unroll
      for (int r = 0; r < 8; ++r) {
        int m = tileM + fm * 16 + rbase + r;
        float v = c[r] + bs;
        if (act == 1) v = fmaxf(v, 0.f);
        if (outF) outFg[(size_t)m * ldo + n] = v;
        else      outHg[(size_t)m * ldo + n] = (bf16)v;
      }
    }
  }
}

// --- LSTM elementwise update (PyTorch i,f,g,o gate order) -------------------
__global__ void lstm_update_kernel(const float* __restrict__ gates,
                                   float* __restrict__ c,
                                   bf16* __restrict__ Hbf,
                                   bf16* __restrict__ hs_t)
{
  int idx = blockIdx.x * blockDim.x + threadIdx.x;   // B_*H_ threads
  int b = idx >> 9;
  int j = idx & (H_ - 1);
  const float* g = gates + (size_t)b * G4H;
  float gi = g[j];
  float gf = g[j + H_];
  float gg = g[j + 2 * H_];
  float go = g[j + 3 * H_];
  float si = 1.f / (1.f + __expf(-gi));
  float sf = 1.f / (1.f + __expf(-gf));
  float so = 1.f / (1.f + __expf(-go));
  float cn = sf * c[idx] + si * tanhf(gg);
  c[idx] = cn;
  float h = so * tanhf(cn);
  Hbf[idx]  = (bf16)h;
  hs_t[idx] = (bf16)h;
}

// --- small final projections: wave-per-row dot(1024) ------------------------
__global__ void head_out_kernel(const bf16* __restrict__ h2,
                                const float* __restrict__ W,
                                const float* __restrict__ bias,
                                float* __restrict__ out,
                                int N, int nout, int dotanh)
{
  int wid = threadIdx.x >> 5, lane = threadIdx.x & 31;
  int row = blockIdx.x * 4 + wid;
  if (row >= N) return;
  const bf16* hr = h2 + (size_t)row * M_;
  for (int j = 0; j < nout; ++j) {
    const float* w = W + (size_t)j * M_;
    float s = 0.f;
    for (int k = lane; k < M_; k += 32)
      s += (float)hr[k] * w[k];
#pragma unroll
    for (int off = 16; off > 0; off >>= 1)
      s += __shfl_xor(s, off, 32);
    if (lane == 0) {
      float v = s + bias[j];
      if (dotanh) v = tanhf(v);
      out[(size_t)row * nout + j] = v;
    }
  }
}

// --- prep kernels ------------------------------------------------------------
__global__ void cvt_bf16_kernel(const float* __restrict__ in, bf16* __restrict__ out, long n) {
  long i = (long)blockIdx.x * blockDim.x + threadIdx.x;
  long stride = (long)gridDim.x * blockDim.x;
  for (; i < n; i += stride) out[i] = (bf16)in[i];
}

__global__ void pad_wih_kernel(const float* __restrict__ Wih, bf16* __restrict__ outp) {
  int idx = blockIdx.x * blockDim.x + threadIdx.x;   // G4H*EGOP
  int n = idx >> 5, k = idx & 31;
  float v = (k < 12) ? Wih[n * 12 + k] : 0.f;
  outp[idx] = (bf16)v;
}

__global__ void build_ego_kernel(const float* __restrict__ state, bf16* __restrict__ ego) {
  int idx = blockIdx.x * blockDim.x + threadIdx.x;   // T_*B_*EGOP
  int k = idx & 31;
  long tb = idx >> 5;
  float v = (k < 12) ? state[tb * S_ + k] : 0.f;
  ego[idx] = (bf16)v;
}

__global__ void seq_scan_kernel(const int* __restrict__ len, int* __restrict__ offsets) {
  __shared__ int sh[B_];
  int t = threadIdx.x;
  sh[t] = len[t];
  __syncthreads();
  if (t == 0) {
    int acc = 0;
    for (int i = 0; i < B_; ++i) { int l = sh[i]; sh[i] = acc; acc += l; }
  }
  __syncthreads();
  offsets[t] = sh[t];
}

__global__ void fill_i32_kernel(int* __restrict__ p, int n, int val) {
  int i = blockIdx.x * blockDim.x + threadIdx.x;
  if (i < n) p[i] = val;
}

__global__ void fill_rowmap_kernel(const int* __restrict__ len,
                                   const int* __restrict__ offsets,
                                   int* __restrict__ rowmap) {
  int b = blockIdx.x;
  int l = len[b], o = offsets[b];
  for (int t = threadIdx.x; t < l; t += blockDim.x)
    rowmap[o + t] = (b << 16) | t;
}

// ---------------------------------------------------------------------------
extern "C" void kernel_launch(void* const* d_in, const int* in_sizes, int n_in,
                              void* d_out, int out_size, void* d_ws, size_t ws_size,
                              hipStream_t stream)
{
  (void)in_sizes; (void)n_in; (void)ws_size;
  const float* state  = (const float*)d_in[0];
  const float* hidden = (const float*)d_in[1];
  const float* cell   = (const float*)d_in[2];
  const int*   lens   = (const int*)d_in[3];
  const float* W_ih   = (const float*)d_in[4];
  const float* W_hh   = (const float*)d_in[5];
  const float* b_ih   = (const float*)d_in[6];
  const float* b_hh   = (const float*)d_in[7];
  const float* aW1    = (const float*)d_in[8];
  const float* ab1    = (const float*)d_in[9];
  const float* aW2    = (const float*)d_in[10];
  const float* ab2    = (const float*)d_in[11];
  const float* locW   = (const float*)d_in[12];
  const float* locb   = (const float*)d_in[13];
  const float* cW1    = (const float*)d_in[14];
  const float* cb1    = (const float*)d_in[15];
  const float* cW2    = (const float*)d_in[16];
  const float* cb2    = (const float*)d_in[17];
  const float* cW3    = (const float*)d_in[18];
  const float* cb3    = (const float*)d_in[19];
  float* out = (float*)d_out;

  const int N    = out_size / 3;            // ragged row count (act: 2N, v: N)
  const int Npad = (N + 127) & ~127;

  // --- workspace carve-out ---
  size_t off = 0;
  auto alloc = [&](size_t bytes) -> void* {
    void* p = (char*)d_ws + off;
    off = (off + bytes + 255) & ~(size_t)255;
    return p;
  };
  bf16* Whh_bf   = (bf16*)alloc((size_t)G4H * H_ * 2);
  bf16* Wih_pad  = (bf16*)alloc((size_t)G4H * EGOP * 2);
  bf16* ego      = (bf16*)alloc((size_t)T_ * B_ * EGOP * 2);
  bf16* state_bf = (bf16*)alloc((size_t)T_ * B_ * S_ * 2);
  bf16* hs       = (bf16*)alloc((size_t)T_ * B_ * H_ * 2);
  bf16* Hbf      = (bf16*)alloc((size_t)B_ * H_ * 2);
  float* cbuf    = (float*)alloc((size_t)B_ * H_ * 4);
  float* gates   = (float*)alloc((size_t)B_ * G4H * 4);
  bf16* W1a_bf   = (bf16*)alloc((size_t)M_ * (S_ + H_) * 2);
  bf16* W2a_bf   = (bf16*)alloc((size_t)M_ * M_ * 2);
  bf16* W1c_bf   = (bf16*)alloc((size_t)M_ * (S_ + H_) * 2);
  bf16* W2c_bf   = (bf16*)alloc((size_t)M_ * M_ * 2);
  int*  offsets  = (int*)alloc((size_t)B_ * 4);
  int*  rowmap   = (int*)alloc((size_t)Npad * 4);
  bf16* h1       = (bf16*)alloc((size_t)Npad * M_ * 2);
  bf16* h2       = (bf16*)alloc((size_t)Npad * M_ * 2);

  // --- precision conversion (weights L2-resident afterwards) ---
  cvt_bf16_kernel<<<2048, 256, 0, stream>>>(W_hh, Whh_bf, (long)G4H * H_);
  cvt_bf16_kernel<<<2048, 256, 0, stream>>>(aW1, W1a_bf, (long)M_ * (S_ + H_));
  cvt_bf16_kernel<<<2048, 256, 0, stream>>>(aW2, W2a_bf, (long)M_ * M_);
  cvt_bf16_kernel<<<2048, 256, 0, stream>>>(cW1, W1c_bf, (long)M_ * (S_ + H_));
  cvt_bf16_kernel<<<2048, 256, 0, stream>>>(cW2, W2c_bf, (long)M_ * M_);
  cvt_bf16_kernel<<<2048, 256, 0, stream>>>(hidden, Hbf, (long)B_ * H_);
  cvt_bf16_kernel<<<4096, 256, 0, stream>>>(state, state_bf, (long)T_ * B_ * S_);
  pad_wih_kernel<<<(G4H * EGOP) / 256, 256, 0, stream>>>(W_ih, Wih_pad);
  build_ego_kernel<<<(T_ * B_ * EGOP) / 256, 256, 0, stream>>>(state, ego);
  hipMemcpyAsync(cbuf, cell, (size_t)B_ * H_ * 4, hipMemcpyDeviceToDevice, stream);

  // --- ragged index map ---
  fill_i32_kernel<<<(Npad + 255) / 256, 256, 0, stream>>>(rowmap, Npad, 0);
  seq_scan_kernel<<<1, B_, 0, stream>>>(lens, offsets);
  fill_rowmap_kernel<<<B_, 128, 0, stream>>>(lens, offsets, rowmap);

  // --- LSTM recurrence: 128 steps of [256x544]x[544x2048] WMMA GEMM + update
  for (int t = 0; t < T_; ++t) {
    const bf16* egot = ego + (size_t)t * B_ * EGOP;
    wmma_gemm_kernel<<<dim3(B_ / 128, G4H / 64), 256, 0, stream>>>(
        /*mode=*/0,
        Hbf, H_, H_,
        egot, EGOP, EGOP,
        Whh_bf, H_,
        Wih_pad, EGOP,
        b_ih, b_hh,
        gates, nullptr, G4H,
        /*act=*/0, nullptr);
    lstm_update_kernel<<<(B_ * H_) / 256, 256, 0, stream>>>(
        gates, cbuf, Hbf, hs + (size_t)t * B_ * H_);
  }

  // --- heads: x = [state | h_t] gathered via rowmap, two relu GEMMs, tiny proj
  auto run_head = [&](const bf16* W1b, const float* bias1v,
                      const bf16* W2b, const float* bias2v,
                      const float* Wout, const float* bout,
                      int nout, int dotanh, float* outp) {
    wmma_gemm_kernel<<<dim3(Npad / 128, M_ / 64), 256, 0, stream>>>(
        /*mode=*/1,
        state_bf, 0, S_,
        hs, 0, H_,
        W1b, S_ + H_,
        W1b + S_, S_ + H_,
        bias1v, nullptr,
        nullptr, h1, M_,
        /*act=*/1, rowmap);
    wmma_gemm_kernel<<<dim3(Npad / 128, M_ / 64), 256, 0, stream>>>(
        /*mode=*/2,
        h1, M_, M_,
        nullptr, 0, 0,
        W2b, M_,
        nullptr, 0,
        bias2v, nullptr,
        nullptr, h2, M_,
        /*act=*/1, rowmap);
    head_out_kernel<<<(N + 3) / 4, 128, 0, stream>>>(h2, Wout, bout, outp, N, nout, dotanh);
  };

  run_head(W1a_bf, ab1, W2a_bf, ab2, locW, locb, 2, 1, out);                 // actor
  run_head(W1c_bf, cb1, W2c_bf, cb2, cW3, cb3, 1, 0, out + (size_t)2 * N);   // critic
}
